// EmbedMatcher_LSTMAE_26843545600085
// MI455X (gfx1250) — matrix operand order repaired and checked by reading, not compiled, not run on hardware
//
#include <hip/hip_runtime.h>
#include <hip/hip_bf16.h>
#include <math.h>

// ---------------- types for WMMA ----------------
typedef __attribute__((ext_vector_type(16))) __bf16 v16bf;
typedef __attribute__((ext_vector_type(8)))  __bf16 v8bf;
typedef __attribute__((ext_vector_type(8)))  float  v8f;
typedef __attribute__((ext_vector_type(4)))  float  v4f;

#define WMMA_BF16(A, Bm, C) \
  __builtin_amdgcn_wmma_f32_16x16x32_bf16(false, (A), false, (Bm), (short)0, (C), false, false)

// ---------------- problem constants ----------------
#define VOCAB 200000
#define DD    128
#define BB    256
#define PP    64
#define RR    32
#define M1    (BB * PP * 3)   // 49152 rows for the D-GEMMs
#define MG    (BB * PP)       // 16384 rows for the GRU
#define EPS   1e-8f

// ---------------- helpers ----------------
__device__ __forceinline__ float wave_sum(float v) {
#pragma unroll
  for (int o = 16; o > 0; o >>= 1) v += __shfl_xor(v, o, 32);
  return v;  // full sum in all lanes (wave32)
}

// A fragment (16x32 bf16): lane = hi*16 + m ; element e<8 -> K = kb+8*hi+e,
// e>=8 -> K = kb+16+8*hi+(e-8). Two contiguous 16B runs.
__device__ __forceinline__ v16bf load_a_bf16(const __bf16* row, int kb, int hi) {
  union { v16bf v; v8bf h[2]; } u;
  u.h[0] = *(const v8bf*)(row + kb + 8 * hi);
  u.h[1] = *(const v8bf*)(row + kb + 16 + 8 * hi);
  return u.v;
}

__device__ __forceinline__ v16bf load_a_f32(const float* row, int kb, int hi) {
  union { v16bf v; __bf16 e[16]; } u;
  v4f f0 = *(const v4f*)(row + kb + 8 * hi);
  v4f f1 = *(const v4f*)(row + kb + 8 * hi + 4);
  v4f f2 = *(const v4f*)(row + kb + 16 + 8 * hi);
  v4f f3 = *(const v4f*)(row + kb + 16 + 8 * hi + 4);
#pragma unroll
  for (int i = 0; i < 4; ++i) {
    u.e[i]      = (__bf16)f0[i];
    u.e[4 + i]  = (__bf16)f1[i];
    u.e[8 + i]  = (__bf16)f2[i];
    u.e[12 + i] = (__bf16)f3[i];
  }
  return u.v;
}

// B fragment (32x16 bf16): lane = hi*16 + n ; element e<8 -> K = kb+8*hi+e,
// e>=8 -> K = kb+16+8*hi+(e-8). Two 16B runs of W[n][*] (row-major W gives W^T columns for free).
__device__ __forceinline__ v16bf load_b_bf16(const __bf16* wrow, int kb, int hi) {
  union { v16bf v; v8bf h[2]; } u;
  u.h[0] = *(const v8bf*)(wrow + kb + 8 * hi);
  u.h[1] = *(const v8bf*)(wrow + kb + 16 + 8 * hi);
  return u.v;
}

__device__ __forceinline__ float sigmoidf_(float x) { return 1.0f / (1.0f + expf(-x)); }

// ---------------- kernel 0: all f32 -> bf16 weight conversions, one launch ---
__global__ void k_cvt_all(const float* __restrict__ wd, const float* __restrict__ we,
                          const float* __restrict__ wp, const float* __restrict__ wih,
                          const float* __restrict__ whh,
                          __bf16* __restrict__ wd_o, __bf16* __restrict__ we_o,
                          __bf16* __restrict__ wp_o, __bf16* __restrict__ wih_o,
                          __bf16* __restrict__ whh_o) {
  int i = blockIdx.x * 256 + threadIdx.x;
  const int SQ = DD * DD;          // 16384
  const int G  = 3 * DD * DD;      // 49152
  if (i < SQ) {
    wd_o[i] = (__bf16)wd[i];
    we_o[i] = (__bf16)we[i];
    wp_o[i] = (__bf16)wp[i];
  }
  if (i < G) {
    wih_o[i] = (__bf16)wih[i];
    whh_o[i] = (__bf16)whh[i];
  }
}

// ---------------- kernel 1: per-batch stats (hts, 1/|h|, 1/|t|, 1/sum_exp) ----
__global__ void k_batch_stats(const int* __restrict__ pair,
                              const int* __restrict__ relset,
                              const float* __restrict__ emb,
                              const float* __restrict__ euw,
                              const float* __restrict__ eub,
                              float* __restrict__ hts,
                              float* __restrict__ inv_nh,
                              float* __restrict__ inv_nt,
                              float* __restrict__ inv_se) {
  int b = blockIdx.x;
  int tid = threadIdx.x;            // 128 threads
  int lane = tid & 31, w = tid >> 5;
  __shared__ float sdot[3];
  __shared__ float ssum;
  if (tid < 3) sdot[tid] = 0.0f;
  if (tid == 0) ssum = 0.0f;
  __syncthreads();

  const float* hrow = emb + (size_t)pair[b * 2 + 0] * DD;
  const float* trow = emb + (size_t)pair[b * 2 + 1] * DD;
  float h = hrow[tid], t = trow[tid];
  float p0 = wave_sum(h * h);
  float p1 = wave_sum(t * t);
  float p2 = wave_sum(h * t);
  if (lane == 0) {
    atomicAdd(&sdot[0], p0);
    atomicAdd(&sdot[1], p1);
    atomicAdd(&sdot[2], p2);
  }
  // sum_exp over relation set: wave w handles 8 rows; lane covers 4 dims.
  v4f eu = *(const v4f*)(euw + lane * 4);
  float acc = 0.0f;
  for (int r = w * 8; r < w * 8 + 8; ++r) {
    const float* rs = emb + (size_t)relset[b * RR + r] * DD;
    v4f rv = *(const v4f*)(rs + lane * 4);
    float s = rv[0] * eu[0] + rv[1] * eu[1] + rv[2] * eu[2] + rv[3] * eu[3];
    s = wave_sum(s);
    acc += expf(s + eub[0]);
  }
  if (lane == 0) atomicAdd(&ssum, acc);
  __syncthreads();
  if (tid == 0) {
    float nh = fmaxf(sqrtf(sdot[0]), EPS);
    float nt = fmaxf(sqrtf(sdot[1]), EPS);
    inv_nh[b] = 1.0f / nh;
    inv_nt[b] = 1.0f / nt;
    hts[b]    = sdot[2] / (nh * nt);
    inv_se[b] = 1.0f / ssum;
  }
}

// ---------------- kernel 2: per-(b,p) sims + build d_in/e_in (bf16) ----------
__global__ void k_build_inputs(const int* __restrict__ path,
                               const int* __restrict__ pair,
                               const float* __restrict__ emb,
                               const float* __restrict__ euw,
                               const float* __restrict__ eub,
                               const float* __restrict__ hts,
                               const float* __restrict__ inv_nh,
                               const float* __restrict__ inv_nt,
                               const float* __restrict__ inv_se,
                               __bf16* __restrict__ d_in,
                               __bf16* __restrict__ e_in) {
  int lane = threadIdx.x & 31;
  int gw = blockIdx.x * (blockDim.x >> 5) + (threadIdx.x >> 5);  // one wave per (b,p)
  int b = gw / PP;

  v4f hv = *(const v4f*)(emb + (size_t)pair[b * 2 + 0] * DD + lane * 4);
  v4f tv = *(const v4f*)(emb + (size_t)pair[b * 2 + 1] * DD + lane * 4);
  v4f eu = *(const v4f*)(euw + lane * 4);

  v4f ev[3];
  float dhe[3], dte[3], due[3], nrm[3];
#pragma unroll
  for (int j = 0; j < 3; ++j) {
    int idx = path[gw * 3 + j];
    ev[j] = *(const v4f*)(emb + (size_t)idx * DD + lane * 4);
    float ph = 0, pt = 0, pu = 0, pn = 0;
#pragma unroll
    for (int i = 0; i < 4; ++i) {
      float e = ev[j][i];
      ph += hv[i] * e; pt += tv[i] * e; pu += eu[i] * e; pn += e * e;
    }
    dhe[j] = wave_sum(ph);
    dte[j] = wave_sum(pt);
    due[j] = wave_sum(pu);
    nrm[j] = wave_sum(pn);
  }
  float inh = inv_nh[b], intl = inv_nt[b], hb = hts[b], ise = inv_se[b];
  float in1 = 1.0f / fmaxf(sqrtf(nrm[0]), EPS);
  float in2 = 1.0f / fmaxf(sqrtf(nrm[1]), EPS);
  float sim1 = 0.5f * (dhe[0] * inh * in1 + dte[0] * intl * in1);
  float sim2 = 0.5f * (dhe[1] * inh * in2 + dte[1] * intl * in2);
  float sr0 = 0.5f * (hb + sim1);
  float sr1 = 0.5f * (sim1 + sim2);
  float sr2 = 0.5f * (hb + sim2);
  float sd[3] = { (1.0f - sr0) * 0.5f, (1.0f - sr1) * 0.5f, (1.0f - sr2) * 0.5f };
  float eb0 = eub[0];
  float es[3] = { expf(due[0] + eb0) * ise, expf(due[1] + eb0) * ise, expf(due[2] + eb0) * ise };

#pragma unroll
  for (int j = 0; j < 3; ++j) {
    size_t row = (size_t)(gw * 3 + j) * DD + lane * 4;
#pragma unroll
    for (int i = 0; i < 4; ++i) {
      float e = ev[j][i];
      d_in[row + i] = (__bf16)(e * sd[j]);
      e_in[row + i] = (__bf16)(e * es[j]);
    }
  }
}

// ---------------- kernel 3: fused cost GEMMs + Wp GEMM (WMMA) ----------------
// Per block: 64 rows (4 row-tiles); wave w owns output cols [16w, 16w+16).
// B fragments hoisted per K-step and reused across the 4 row-tiles.
__global__ __launch_bounds__(256) void k_gemm_cost_p(
    const __bf16* __restrict__ Ad, const __bf16* __restrict__ Ae,
    const __bf16* __restrict__ Wd, const __bf16* __restrict__ We,
    const __bf16* __restrict__ Wp,
    const float* __restrict__ bd, const float* __restrict__ be,
    const float* __restrict__ bp,
    const int* __restrict__ path_flat, const float* __restrict__ emb,
    __bf16* __restrict__ relin) {
  const int w = threadIdx.x >> 5, lane = threadIdx.x & 31;
  const int ln = lane & 15, hi = lane >> 4;
  const int rowbase = blockIdx.x * 64;

  __shared__ __bf16 a2[64][DD];
  __shared__ int sidx[64];
  if (threadIdx.x < 64) sidx[threadIdx.x] = path_flat[rowbase + threadIdx.x];
  __syncthreads();

  const int n = w * 16 + ln;  // output column (0..127)
  const __bf16* wdp = Wd + (size_t)n * DD;
  const __bf16* wep = We + (size_t)n * DD;

  v8f acc[4] = {{}, {}, {}, {}};
#pragma unroll
  for (int kk = 0; kk < 4; ++kk) {
    const int kb = kk * 32;
    v16bf bD = load_b_bf16(wdp, kb, hi);
    v16bf bE = load_b_bf16(wep, kb, hi);
#pragma unroll
    for (int rt = 0; rt < 4; ++rt) {
      const size_t arow = (size_t)(rowbase + rt * 16 + ln) * DD;
      acc[rt] = WMMA_BF16(load_a_bf16(Ad + arow, kb, hi), bD, acc[rt]);
      acc[rt] = WMMA_BF16(load_a_bf16(Ae + arow, kb, hi), bE, acc[rt]);
    }
  }
  const float bias = bd[n] + be[n];
#pragma unroll
  for (int rt = 0; rt < 4; ++rt) {
#pragma unroll
    for (int r = 0; r < 8; ++r) {
      int mm = rt * 16 + r + 8 * hi;  // local row within 64-row block
      float relv = emb[(size_t)sidx[mm] * DD + n];
      float c = acc[rt][r] + bias;
      c = c > 0.0f ? c : 0.0f;
      a2[mm][n] = (__bf16)(relv - 0.001f * c);
    }
  }
  __syncthreads();

  v8f acc2[4] = {{}, {}, {}, {}};
  const __bf16* wpp = Wp + (size_t)n * DD;
#pragma unroll
  for (int kk = 0; kk < 4; ++kk) {
    const int kb = kk * 32;
    v16bf bP = load_b_bf16(wpp, kb, hi);
#pragma unroll
    for (int rt = 0; rt < 4; ++rt) {
      acc2[rt] = WMMA_BF16(load_a_bf16(&a2[rt * 16 + ln][0], kb, hi), bP, acc2[rt]);
    }
  }
  const float bpv = bp[n];
#pragma unroll
  for (int rt = 0; rt < 4; ++rt) {
#pragma unroll
    for (int r = 0; r < 8; ++r) {
      int mm = rt * 16 + r + 8 * hi;
      relin[(size_t)(rowbase + mm) * DD + n] = (__bf16)(acc2[rt][r] + bpv);
    }
  }
}

// ---------------- kernel 4: one GRU step (WMMA for gi/gh) --------------------
// Per block: 32 rows (2 row-tiles); wave w owns D-columns [16w,16w+16) of all
// 3 gates. B fragments hoisted per K-step, reused across the 2 row-tiles.
// first=1: h == 0, so skip the h GEMM and the hprev term entirely.
__global__ __launch_bounds__(256) void k_gru_step(
    const __bf16* __restrict__ X,      // rel_in: (MG*3) x D, row = bp*3 + t
    const float* __restrict__ Hin,     // MG x D (ignored when first=1)
    const __bf16* __restrict__ Wih,    // 384 x 128 bf16
    const __bf16* __restrict__ Whh,    // 384 x 128 bf16
    const float* __restrict__ bih, const float* __restrict__ bhh,
    int t, int first, float* __restrict__ Hout) {
  const int w = threadIdx.x >> 5, lane = threadIdx.x & 31;
  const int ln = lane & 15, hi = lane >> 4;
  const int rowbase = blockIdx.x * 32;
  const int nD = w * 16 + ln;  // output column within D

  v8f gi[2][3] = {{{}, {}, {}}, {{}, {}, {}}};
  v8f gh[2][3] = {{{}, {}, {}}, {{}, {}, {}}};
#pragma unroll
  for (int kk = 0; kk < 4; ++kk) {
    const int kb = kk * 32;
    v16bf bi0 = load_b_bf16(Wih + (size_t)(0   + nD) * DD, kb, hi);
    v16bf bi1 = load_b_bf16(Wih + (size_t)(128 + nD) * DD, kb, hi);
    v16bf bi2 = load_b_bf16(Wih + (size_t)(256 + nD) * DD, kb, hi);
#pragma unroll
    for (int rt = 0; rt < 2; ++rt) {
      const int bp = rowbase + rt * 16 + ln;
      v16bf ax = load_a_bf16(X + (size_t)(bp * 3 + t) * DD, kb, hi);
      gi[rt][0] = WMMA_BF16(ax, bi0, gi[rt][0]);
      gi[rt][1] = WMMA_BF16(ax, bi1, gi[rt][1]);
      gi[rt][2] = WMMA_BF16(ax, bi2, gi[rt][2]);
    }
    if (!first) {
      v16bf bh0 = load_b_bf16(Whh + (size_t)(0   + nD) * DD, kb, hi);
      v16bf bh1 = load_b_bf16(Whh + (size_t)(128 + nD) * DD, kb, hi);
      v16bf bh2 = load_b_bf16(Whh + (size_t)(256 + nD) * DD, kb, hi);
#pragma unroll
      for (int rt = 0; rt < 2; ++rt) {
        const int bp = rowbase + rt * 16 + ln;
        v16bf ah = load_a_f32(Hin + (size_t)bp * DD, kb, hi);
        gh[rt][0] = WMMA_BF16(ah, bh0, gh[rt][0]);
        gh[rt][1] = WMMA_BF16(ah, bh1, gh[rt][1]);
        gh[rt][2] = WMMA_BF16(ah, bh2, gh[rt][2]);
      }
    }
  }
  const float bir = bih[nD], biz = bih[128 + nD], bin = bih[256 + nD];
  const float bhr = bhh[nD], bhz = bhh[128 + nD], bhn = bhh[256 + nD];
#pragma unroll
  for (int rt = 0; rt < 2; ++rt) {
#pragma unroll
    for (int r = 0; r < 8; ++r) {
      int mm = rt * 16 + r + 8 * hi;
      float hprev = first ? 0.0f : Hin[(size_t)(rowbase + mm) * DD + nD];
      float rg = sigmoidf_(gi[rt][0][r] + bir + gh[rt][0][r] + bhr);
      float zg = sigmoidf_(gi[rt][1][r] + biz + gh[rt][1][r] + bhz);
      float ng = tanhf(gi[rt][2][r] + bin + rg * (gh[rt][2][r] + bhn));
      Hout[(size_t)(rowbase + mm) * DD + nD] = (1.0f - zg) * ng + zg * hprev;
    }
  }
}

// ---------------- host side ----------------
extern "C" void kernel_launch(void* const* d_in, const int* in_sizes, int n_in,
                              void* d_out, int out_size, void* d_ws, size_t ws_size,
                              hipStream_t stream) {
  const int*   path    = (const int*)d_in[0];    // (B,P,3)
  const int*   pair    = (const int*)d_in[1];    // (B,2)
  const int*   relset  = (const int*)d_in[2];    // (B,R)
  const float* emb     = (const float*)d_in[3];  // (VOCAB,D)
  const float* w_d_w   = (const float*)d_in[4];
  const float* w_d_b   = (const float*)d_in[5];
  const float* w_e_w   = (const float*)d_in[6];
  const float* w_e_b   = (const float*)d_in[7];
  const float* w_p_w   = (const float*)d_in[8];
  const float* w_p_b   = (const float*)d_in[9];
  const float* e_u_w   = (const float*)d_in[10];
  const float* e_u_b   = (const float*)d_in[11];
  const float* gru_wih = (const float*)d_in[12];
  const float* gru_whh = (const float*)d_in[13];
  const float* gru_bih = (const float*)d_in[14];
  const float* gru_bhh = (const float*)d_in[15];

  char* ws = (char*)d_ws;
  size_t off = 0;
  auto take = [&](size_t bytes) -> void* {
    void* p = ws + off;
    off = (off + bytes + 255) & ~(size_t)255;
    return p;
  };
  __bf16* wd_bf  = (__bf16*)take(DD * DD * sizeof(__bf16));
  __bf16* we_bf  = (__bf16*)take(DD * DD * sizeof(__bf16));
  __bf16* wp_bf  = (__bf16*)take(DD * DD * sizeof(__bf16));
  __bf16* wih_bf = (__bf16*)take(3 * DD * DD * sizeof(__bf16));
  __bf16* whh_bf = (__bf16*)take(3 * DD * DD * sizeof(__bf16));
  float*  hts    = (float*)take(BB * sizeof(float));
  float*  inv_nh = (float*)take(BB * sizeof(float));
  float*  inv_nt = (float*)take(BB * sizeof(float));
  float*  inv_se = (float*)take(BB * sizeof(float));
  __bf16* d_in_m = (__bf16*)take((size_t)M1 * DD * sizeof(__bf16));
  __bf16* e_in_m = (__bf16*)take((size_t)M1 * DD * sizeof(__bf16));
  __bf16* relin  = (__bf16*)take((size_t)M1 * DD * sizeof(__bf16));
  float*  h0     = (float*)take((size_t)MG * DD * sizeof(float));
  float*  h1     = (float*)take((size_t)MG * DD * sizeof(float));
  (void)ws_size; (void)n_in; (void)in_sizes; (void)out_size;

  // 0) all weight conversions, one launch (sized by the largest: 3*D*D)
  k_cvt_all<<<(3 * DD * DD + 255) / 256, 256, 0, stream>>>(
      w_d_w, w_e_w, w_p_w, gru_wih, gru_whh,
      wd_bf, we_bf, wp_bf, wih_bf, whh_bf);

  // 1) per-batch stats
  k_batch_stats<<<BB, 128, 0, stream>>>(pair, relset, emb, e_u_w, e_u_b,
                                        hts, inv_nh, inv_nt, inv_se);

  // 2) build d_in / e_in (one wave per (b,p); 8 waves per block)
  k_build_inputs<<<(BB * PP) / 8, 256, 0, stream>>>(path, pair, emb, e_u_w, e_u_b,
                                                    hts, inv_nh, inv_nt, inv_se,
                                                    d_in_m, e_in_m);

  // 3) fused cost GEMMs + Wp GEMM -> rel_in (64 rows per block)
  k_gemm_cost_p<<<M1 / 64, 256, 0, stream>>>(d_in_m, e_in_m, wd_bf, we_bf, wp_bf,
                                             w_d_b, w_e_b, w_p_b, path, emb, relin);

  // 4) GRU, 3 steps; step 0 skips the zero-h GEMM; last step writes d_out
  float* out = (float*)d_out;
  k_gru_step<<<MG / 32, 256, 0, stream>>>(relin, h0, wih_bf, whh_bf, gru_bih, gru_bhh, 0, 1, h1);
  k_gru_step<<<MG / 32, 256, 0, stream>>>(relin, h1, wih_bf, whh_bf, gru_bih, gru_bhh, 1, 0, h0);
  k_gru_step<<<MG / 32, 256, 0, stream>>>(relin, h0, wih_bf, whh_bf, gru_bih, gru_bhh, 2, 0, out);
}
